// TransModel_48576080118169
// MI455X (gfx1250) — matrix operand order repaired and checked by reading, not compiled, not run on hardware
//
#include <hip/hip_runtime.h>
#include <hip/hip_bf16.h>
#include <math.h>

// ---------------- problem constants (match reference) ----------------
#define Bc    32
#define Sc    64
#define Lc    64
#define EMBc  256
#define HIDc  512
#define ATTc  512
#define VOCc  32000
#define TSZ   1280            // 2*HID + EMB
#define OUTK  1536            // TSZ + EMB
#define G3    1536            // 3*HID
#define D3    3840            // 3*TSIZE
#define SOSid 1

typedef __attribute__((ext_vector_type(16))) __bf16 v16bf;
typedef __attribute__((ext_vector_type(8)))  __bf16 v8bf;
typedef __attribute__((ext_vector_type(8)))  float  v8f;
typedef __attribute__((__vector_size__(4 * sizeof(int)))) int v4i;

// gfx1250 async global->LDS path (guarded; falls back to load+ds_store)
#if defined(__has_builtin)
# if __has_builtin(__builtin_amdgcn_global_load_async_to_lds_b128) && \
     __has_builtin(__builtin_amdgcn_s_wait_asynccnt)
#  define USE_ASYNC_LDS 1
# endif
#endif
#ifndef USE_ASYNC_LDS
# define USE_ASYNC_LDS 0
#endif

__device__ __forceinline__ float sigf(float x) { return 1.0f / (1.0f + __expf(-x)); }

__device__ __forceinline__ v16bf pack_a(const __bf16* p) {
  v8bf lo = *(const v8bf*)p;
  v8bf hi = *(const v8bf*)(p + 16);
  v16bf a;
#pragma unroll
  for (int j = 0; j < 8; ++j) { a[j] = lo[j]; a[j + 8] = hi[j]; }
  return a;
}

// ====================================================================
// Small GEMM (latency-optimized, one 16x16 tile per wave):
//   C[M,N] = act( A[M,K] * W[N,K]^T + bias[N] ), bf16 in / f32 accum.
// Used for the M=32 recurrent steps where wave parallelism matters.
// ====================================================================
__global__ void gemm_small_wmma(const __bf16* __restrict__ A, int lda,
                                const __bf16* __restrict__ W, int ldw,
                                const float* __restrict__ bias,
                                float* __restrict__ C, __bf16* __restrict__ Cbf,
                                int ldc, int M, int N, int K, int act) {
  const int gtid  = blockIdx.x * blockDim.x + threadIdx.x;
  const int wave  = gtid >> 5;
  const int lane  = threadIdx.x & 31;
  const int ntile = N >> 4;
  const int total = (M >> 4) * ntile;
  if (wave >= total) return;                    // wave-uniform: EXEC stays full
  const int tm   = wave / ntile;
  const int tn   = wave % ntile;
  const int half = lane >> 4;
  const int l16  = lane & 15;

  const __bf16* __restrict__ Arow = A + (size_t)(tm * 16 + l16) * lda + half * 16;
  const __bf16* __restrict__ Wrow = W + (size_t)(tn * 16 + l16) * ldw + half * 16;

  v8f acc = {};
  for (int k0 = 0; k0 < K; k0 += 32) {
    __builtin_prefetch(Wrow + k0 + 64, 0, 0);
    v16bf a = __builtin_shufflevector(*(const v8bf*)(Arow + k0), *(const v8bf*)(Arow + k0 + 8), 0, 1, 2, 3, 4, 5, 6, 7, 8, 9, 10, 11, 12, 13, 14, 15);
    v16bf b = *(const v16bf*)(Wrow + k0);
    acc = __builtin_amdgcn_wmma_f32_16x16x32_bf16(false, a, false, b,
                                                  (short)0, acc, false, false);
  }
  const int n     = tn * 16 + l16;
  const float bs  = bias ? bias[n] : 0.0f;
  const int mbase = tm * 16 + half * 8;
#pragma unroll
  for (int r = 0; r < 8; ++r) {
    float v = acc[r] + bs;
    if (act == 1) v = tanhf(v);
    C[(size_t)(mbase + r) * ldc + n] = v;
    if (Cbf) Cbf[(size_t)(mbase + r) * ldc + n] = (__bf16)v;
  }
}

// ====================================================================
// Big GEMM (throughput-optimized): per wave 32x64 C tile
// (2 M-subtiles x 4 N-subtiles -> 8 WMMAs / K-step).
// A 32-row block is staged once per workgroup into padded LDS
// (async global->LDS when available), shared by all 8 waves.
// MODE 0: C[row*ldc + n]
// MODE 1: final logits; row r = l*B + b -> C[(b*Lc + l)*ldc + n]
// ====================================================================
template <int MODE>
__global__ void gemm_big_wmma(const __bf16* __restrict__ A, int lda,
                              const __bf16* __restrict__ W, int ldw,
                              const float* __restrict__ bias,
                              float* __restrict__ C, int ldc,
                              int M, int N, int K, int act) {
  __shared__ __align__(32) __bf16 Atile[32 * (1536 + 8)];   // K <= 1536, +8 pad
  const int Kp   = K + 8;
  const int tid  = threadIdx.x;
  const int lane = tid & 31;
  const int w    = tid >> 5;                 // wave in block: 0..7
  const int nt64 = N >> 6;                   // 64-col tile count
  const int bpr  = (nt64 + 7) >> 3;          // blocks per 32-row stripe
  const int tmRow = blockIdx.x / bpr;
  const int nblk  = blockIdx.x % bpr;

  // ---- stage A rows [tmRow*32, +32) x [0,K) into LDS ----
  {
    const int cpr   = K >> 3;                // 8-elem (16B) chunks per row
    const int total = 32 * cpr;
    const __bf16* Abase = A + (size_t)tmRow * 32 * lda;
    for (int c = tid; c < total; c += 256) {
      int r = c / cpr, col = (c - r * cpr) * 8;
      const __bf16* g = Abase + (size_t)r * lda + col;
      __bf16* l = &Atile[r * Kp + col];
#if USE_ASYNC_LDS
      __builtin_amdgcn_global_load_async_to_lds_b128(
          (__attribute__((address_space(1))) v4i*)g,
          (__attribute__((address_space(3))) v4i*)l, 0, 0);
#else
      *(v8bf*)l = *(const v8bf*)g;
#endif
    }
#if USE_ASYNC_LDS
    __builtin_amdgcn_s_wait_asynccnt(0);
#endif
    __syncthreads();
  }

  const int tn64 = nblk * 8 + w;
  if (tn64 >= nt64) return;                  // wave-uniform; after barrier
  const int half = lane >> 4;
  const int l16  = lane & 15;

  const __bf16* __restrict__ Wp[4];
#pragma unroll
  for (int ns = 0; ns < 4; ++ns)
    Wp[ns] = W + (size_t)(tn64 * 64 + ns * 16 + l16) * ldw + half * 16;
  const __bf16* __restrict__ A0 = &Atile[(l16)      * Kp + half * 16];
  const __bf16* __restrict__ A1 = &Atile[(16 + l16) * Kp + half * 16];

  v8f acc[2][4] = {};
  for (int k0 = 0; k0 < K; k0 += 32) {
    __builtin_prefetch(Wp[0] + k0 + 128, 0, 0);
    v16bf a0 = __builtin_shufflevector(*(const v8bf*)(A0 + k0), *(const v8bf*)(A0 + k0 + 8), 0, 1, 2, 3, 4, 5, 6, 7, 8, 9, 10, 11, 12, 13, 14, 15);
    v16bf a1 = __builtin_shufflevector(*(const v8bf*)(A1 + k0), *(const v8bf*)(A1 + k0 + 8), 0, 1, 2, 3, 4, 5, 6, 7, 8, 9, 10, 11, 12, 13, 14, 15);
#pragma unroll
    for (int ns = 0; ns < 4; ++ns) {
      v16bf b = *(const v16bf*)(Wp[ns] + k0);
      acc[0][ns] = __builtin_amdgcn_wmma_f32_16x16x32_bf16(false, a0, false, b,
                                                           (short)0, acc[0][ns], false, false);
      acc[1][ns] = __builtin_amdgcn_wmma_f32_16x16x32_bf16(false, a1, false, b,
                                                           (short)0, acc[1][ns], false, false);
    }
  }

#pragma unroll
  for (int ms = 0; ms < 2; ++ms) {
    const int mbase = tmRow * 32 + ms * 16 + half * 8;
#pragma unroll
    for (int ns = 0; ns < 4; ++ns) {
      const int n  = tn64 * 64 + ns * 16 + l16;
      const float bs = bias ? bias[n] : 0.0f;
#pragma unroll
      for (int r = 0; r < 8; ++r) {
        float v = acc[ms][ns][r] + bs;
        if (act == 1) v = tanhf(v);
        const int row = mbase + r;
        if (MODE == 0) {
          C[(size_t)row * ldc + n] = v;
        } else {
          const int b_ = row & (Bc - 1);     // row = l*B + b
          const int l_ = row >> 5;
          C[((size_t)b_ * Lc + l_) * ldc + n] = v;
        }
      }
    }
  }
}

// ---------------- support kernels ----------------
__global__ void f32_to_bf16_k(const float* __restrict__ src, __bf16* __restrict__ dst, int n) {
  int i = blockIdx.x * blockDim.x + threadIdx.x;
  if (i < n) dst[i] = (__bf16)src[i];
}

__global__ void gather_embed_k(const float* __restrict__ E, const int* __restrict__ ids,
                               __bf16* __restrict__ out, int rows) {
  int idx = blockIdx.x * blockDim.x + threadIdx.x;
  if (idx >= rows * EMBc) return;
  int r = idx / EMBc, e = idx % EMBc;
  out[idx] = (__bf16)E[(size_t)ids[r] * EMBc + e];
}

// decoder input embeddings: row r = l*B + b ; id = (l==0) ? SOS : y[b, l]
__global__ void gather_embed_dec_k(const float* __restrict__ Ey, const int* __restrict__ y,
                                   __bf16* __restrict__ out) {
  int idx = blockIdx.x * blockDim.x + threadIdx.x;
  if (idx >= Lc * Bc * EMBc) return;
  int e = idx % EMBc;
  int r = idx / EMBc;
  int l = r / Bc, b = r % Bc;
  int id = (l == 0) ? SOSid : y[b * Lc + l];
  out[idx] = (__bf16)Ey[(size_t)id * EMBc + e];
}

// fill X_all[:, 1280:1536] with decoder embeddings (known up front)
__global__ void fill_X_emb_k(const __bf16* __restrict__ e_bf, __bf16* __restrict__ X) {
  int idx = blockIdx.x * blockDim.x + threadIdx.x;
  if (idx >= Lc * Bc * EMBc) return;
  int r = idx / EMBc, e = idx % EMBc;
  X[(size_t)r * OUTK + TSZ + e] = e_bf[idx];
}

// encoder GRU gate fusion for one timestep (one direction)
__global__ void gru_enc_pw_k(const float* __restrict__ gi_all, int t,
                             const float* __restrict__ gh,
                             const float* __restrict__ h_prev,
                             float* __restrict__ h_new, __bf16* __restrict__ h_new_bf,
                             __bf16* __restrict__ s_bf, int dir_off) {
  int idx = blockIdx.x * blockDim.x + threadIdx.x;
  if (idx >= Bc * HIDc) return;
  int b = idx / HIDc, j = idx % HIDc;
  const float* gib = gi_all + (size_t)(b * Sc + t) * G3;
  const float* ghb = gh + (size_t)b * G3;
  float r = sigf(gib[j] + ghb[j]);
  float z = sigf(gib[HIDc + j] + ghb[HIDc + j]);
  float n = tanhf(gib[2 * HIDc + j] + r * ghb[2 * HIDc + j]);
  float h = (1.0f - z) * n + z * h_prev[idx];
  h_new[idx]    = h;
  h_new_bf[idx] = (__bf16)h;
  s_bf[(size_t)(b * Sc + t) * (2 * HIDc) + dir_off + j] = (__bf16)h;
}

// attention scores: one wave per (b,s); score = tanh(q+pk) . vW + vb, masked
__global__ void attn_scores_k(const float* __restrict__ q, const float* __restrict__ pk,
                              const float* __restrict__ vW, const float* __restrict__ vb,
                              const unsigned char* __restrict__ mask,
                              float* __restrict__ scores) {
  int gtid = blockIdx.x * blockDim.x + threadIdx.x;
  int wave = gtid >> 5, lane = gtid & 31;
  if (wave >= Bc * Sc) return;
  int b = wave / Sc, s = wave % Sc;
  const float* qb  = q + (size_t)b * ATTc;
  const float* pks = pk + (size_t)(b * Sc + s) * ATTc;
  float acc = 0.0f;
  for (int i = lane; i < ATTc; i += 32) acc += tanhf(qb[i] + pks[i]) * vW[i];
#pragma unroll
  for (int off = 16; off > 0; off >>= 1) acc += __shfl_xor(acc, off, 32);
  if (lane == 0) {
    float sc = acc + vb[0];
    scores[wave] = mask[b * Sc + s] ? sc : -1.0e9f;
  }
}

// softmax over s + context c = sum_s alpha*s[b,s,:] + append e_j -> h_dec
__global__ void attn_ctx_k(const float* __restrict__ scores, const __bf16* __restrict__ s_bf,
                           const __bf16* __restrict__ e_bf,
                           float* __restrict__ hdec_f, __bf16* __restrict__ hdec_bf, int l) {
  __shared__ float alpha[Sc];
  __shared__ float inv_sum;
  int b = blockIdx.x, tid = threadIdx.x;
  if (tid < Sc) alpha[tid] = scores[b * Sc + tid];
  __syncthreads();
  if (tid == 0) {
    float m = -1.0e30f;
    for (int s = 0; s < Sc; ++s) m = fmaxf(m, alpha[s]);
    float sum = 0.0f;
    for (int s = 0; s < Sc; ++s) { float e = __expf(alpha[s] - m); alpha[s] = e; sum += e; }
    inv_sum = 1.0f / sum;
  }
  __syncthreads();
  float inv = inv_sum;
  for (int d = tid; d < TSZ; d += blockDim.x) {
    float v;
    if (d < 2 * HIDc) {
      const __bf16* base = s_bf + (size_t)b * Sc * (2 * HIDc) + d;
      float acc = 0.0f;
      for (int s = 0; s < Sc; ++s) acc += alpha[s] * (float)base[(size_t)s * (2 * HIDc)];
      v = acc * inv;
    } else {
      v = (float)e_bf[(size_t)(l * Bc + b) * EMBc + (d - 2 * HIDc)];
    }
    hdec_f[(size_t)b * TSZ + d]  = v;
    hdec_bf[(size_t)b * TSZ + d] = (__bf16)v;
  }
}

// decoder GRU gates: t_new = (1-z)*n + z*h_dec ; writes t_bf and X_all row
__global__ void gru_dec_pw_k(const float* __restrict__ gi, const float* __restrict__ gh,
                             const float* __restrict__ hdec,
                             __bf16* __restrict__ t_bf, __bf16* __restrict__ X_all, int l) {
  int idx = blockIdx.x * blockDim.x + threadIdx.x;
  if (idx >= Bc * TSZ) return;
  int b = idx / TSZ, j = idx % TSZ;
  const float* gib = gi + (size_t)b * D3;
  const float* ghb = gh + (size_t)b * D3;
  float r = sigf(gib[j] + ghb[j]);
  float z = sigf(gib[TSZ + j] + ghb[TSZ + j]);
  float n = tanhf(gib[2 * TSZ + j] + r * ghb[2 * TSZ + j]);
  float t = (1.0f - z) * n + z * hdec[(size_t)b * TSZ + j];
  t_bf[(size_t)b * TSZ + j]              = (__bf16)t;
  X_all[(size_t)(l * Bc + b) * OUTK + j] = (__bf16)t;
}

// ---------------- host-side helpers ----------------
static inline void launch_gemm_small(hipStream_t st, const __bf16* A, int lda,
                                     const __bf16* W, int ldw, const float* bias,
                                     float* C, __bf16* Cbf, int ldc,
                                     int M, int N, int K, int act) {
  int tiles = (M / 16) * (N / 16);
  int threads = tiles * 32;
  int block = 256, grid = (threads + block - 1) / block;
  gemm_small_wmma<<<grid, block, 0, st>>>(A, lda, W, ldw, bias, C, Cbf, ldc, M, N, K, act);
}

template <int MODE>
static inline void launch_gemm_big(hipStream_t st, const __bf16* A, int lda,
                                   const __bf16* W, int ldw, const float* bias,
                                   float* C, int ldc, int M, int N, int K, int act) {
  int nt64 = N / 64;
  int bpr  = (nt64 + 7) / 8;
  int grid = (M / 32) * bpr;
  gemm_big_wmma<MODE><<<grid, 256, 0, st>>>(A, lda, W, ldw, bias, C, ldc, M, N, K, act);
}

static inline void cvt(hipStream_t st, const float* src, __bf16* dst, int n) {
  f32_to_bf16_k<<<(n + 255) / 256, 256, 0, st>>>(src, dst, n);
}

extern "C" void kernel_launch(void* const* d_in, const int* in_sizes, int n_in,
                              void* d_out, int out_size, void* d_ws, size_t ws_size,
                              hipStream_t stream) {
  (void)in_sizes; (void)n_in; (void)out_size; (void)ws_size;
  // ---- inputs (setup_inputs dict order, params flattened depth-first) ----
  const int*   x        = (const int*)d_in[0];
  const unsigned char* x_mask = (const unsigned char*)d_in[1];
  const float* z        = (const float*)d_in[2];
  const int*   y        = (const int*)d_in[3];
  /* d_in[4] = max_len scalar (== Lc) */
  const float* Ex       = (const float*)d_in[5];
  const float* Ey       = (const float*)d_in[6];
  const float* encW     = (const float*)d_in[7];
  const float* encB     = (const float*)d_in[8];
  const float* decIW    = (const float*)d_in[9];
  const float* decIB    = (const float*)d_in[10];
  const float* fWi      = (const float*)d_in[11];
  const float* fWh      = (const float*)d_in[12];
  const float* fbi      = (const float*)d_in[13];
  const float* fbh      = (const float*)d_in[14];
  const float* bWi      = (const float*)d_in[15];
  const float* bWh      = (const float*)d_in[16];
  const float* bbi      = (const float*)d_in[17];
  const float* bbh      = (const float*)d_in[18];
  const float* dWi      = (const float*)d_in[19];
  const float* dWh      = (const float*)d_in[20];
  const float* dbi      = (const float*)d_in[21];
  const float* dbh      = (const float*)d_in[22];
  const float* akW      = (const float*)d_in[23];
  const float* akB      = (const float*)d_in[24];
  const float* aqW      = (const float*)d_in[25];
  const float* aqB      = (const float*)d_in[26];
  const float* avW      = (const float*)d_in[27];
  const float* avB      = (const float*)d_in[28];
  const float* outW     = (const float*)d_in[29];
  const float* outB     = (const float*)d_in[30];
  float* out = (float*)d_out;

  // ---- workspace carve-up ----
  char* base = (char*)d_ws;
  size_t off = 0;
  auto alloc = [&](size_t bytes) -> void* {
    void* p = base + off;
    off += (bytes + 255) & ~(size_t)255;
    return p;
  };
  __bf16* oW_bf   = (__bf16*)alloc((size_t)VOCc * OUTK * 2);   // 98.3 MB
  __bf16* dWi_bf  = (__bf16*)alloc((size_t)D3 * TSZ * 2);
  __bf16* dWh_bf  = (__bf16*)alloc((size_t)D3 * TSZ * 2);
  __bf16* aqW_bf  = (__bf16*)alloc((size_t)ATTc * TSZ * 2);
  __bf16* akW_bf  = (__bf16*)alloc((size_t)ATTc * 2 * HIDc * 2);
  __bf16* encW_bf = (__bf16*)alloc((size_t)HIDc * HIDc * 2);
  __bf16* decIW_bf= (__bf16*)alloc((size_t)TSZ * HIDc * 2);
  __bf16* fWi_bf  = (__bf16*)alloc((size_t)G3 * EMBc * 2);
  __bf16* fWh_bf  = (__bf16*)alloc((size_t)G3 * HIDc * 2);
  __bf16* bWi_bf  = (__bf16*)alloc((size_t)G3 * EMBc * 2);
  __bf16* bWh_bf  = (__bf16*)alloc((size_t)G3 * HIDc * 2);
  __bf16* z_bf    = (__bf16*)alloc((size_t)Bc * HIDc * 2);
  __bf16* f_bf    = (__bf16*)alloc((size_t)Bc * Sc * EMBc * 2);
  __bf16* e_bf    = (__bf16*)alloc((size_t)Lc * Bc * EMBc * 2);
  float*  gi_f    = (float*)alloc((size_t)Bc * Sc * G3 * 4);   // 12.6 MB
  float*  gi_b    = (float*)alloc((size_t)Bc * Sc * G3 * 4);
  float*  gh_buf  = (float*)alloc((size_t)Bc * G3 * 4);
  float*  h_f     = (float*)alloc((size_t)Bc * HIDc * 4);
  float*  h_b     = (float*)alloc((size_t)Bc * HIDc * 4);
  __bf16* h_f_bf  = (__bf16*)alloc((size_t)Bc * HIDc * 2);
  __bf16* h_b_bf  = (__bf16*)alloc((size_t)Bc * HIDc * 2);
  __bf16* s_bf    = (__bf16*)alloc((size_t)Bc * Sc * 2 * HIDc * 2);  // 4.2 MB
  float*  pk      = (float*)alloc((size_t)Bc * Sc * ATTc * 4);       // 4.2 MB
  float*  t_f32   = (float*)alloc((size_t)Bc * TSZ * 4);             // scratch C
  __bf16* t_bf    = (__bf16*)alloc((size_t)Bc * TSZ * 2);
  float*  q_buf   = (float*)alloc((size_t)Bc * ATTc * 4);
  float*  scores  = (float*)alloc((size_t)Bc * Sc * 4);
  float*  hdec_f  = (float*)alloc((size_t)Bc * TSZ * 4);
  __bf16* hdec_bf = (__bf16*)alloc((size_t)Bc * TSZ * 2);
  float*  gi_dec  = (float*)alloc((size_t)Bc * D3 * 4);
  float*  gh_dec  = (float*)alloc((size_t)Bc * D3 * 4);
  __bf16* X_all   = (__bf16*)alloc((size_t)Lc * Bc * OUTK * 2);      // 6.3 MB

  // ---- 1) weight / activation conversions to bf16 ----
  cvt(stream, outW,  oW_bf,   VOCc * OUTK);
  cvt(stream, dWi,   dWi_bf,  D3 * TSZ);
  cvt(stream, dWh,   dWh_bf,  D3 * TSZ);
  cvt(stream, aqW,   aqW_bf,  ATTc * TSZ);
  cvt(stream, akW,   akW_bf,  ATTc * 2 * HIDc);
  cvt(stream, encW,  encW_bf, HIDc * HIDc);
  cvt(stream, decIW, decIW_bf, TSZ * HIDc);
  cvt(stream, fWi,   fWi_bf,  G3 * EMBc);
  cvt(stream, fWh,   fWh_bf,  G3 * HIDc);
  cvt(stream, bWi,   bWi_bf,  G3 * EMBc);
  cvt(stream, bWh,   bWh_bf,  G3 * HIDc);
  cvt(stream, z,     z_bf,    Bc * HIDc);

  // ---- 2) embeddings ----
  {
    int n = Bc * Sc * EMBc;
    gather_embed_k<<<(n + 255) / 256, 256, 0, stream>>>(Ex, x, f_bf, Bc * Sc);
    gather_embed_dec_k<<<(n + 255) / 256, 256, 0, stream>>>(Ey, y, e_bf);
    fill_X_emb_k<<<(n + 255) / 256, 256, 0, stream>>>(e_bf, X_all);
  }

  // ---- 3) initial states: h0 (both dirs) and t0 ----
  launch_gemm_small(stream, z_bf, HIDc, encW_bf, HIDc, encB, h_f, h_f_bf, HIDc, Bc, HIDc, HIDc, 1);
  launch_gemm_small(stream, z_bf, HIDc, encW_bf, HIDc, encB, h_b, h_b_bf, HIDc, Bc, HIDc, HIDc, 1);
  launch_gemm_small(stream, z_bf, HIDc, decIW_bf, HIDc, decIB, t_f32, t_bf, TSZ, Bc, TSZ, HIDc, 1);

  // ---- 4) hoisted encoder input projections gi = f @ Wi^T + bi (big GEMMs) ----
  launch_gemm_big<0>(stream, f_bf, EMBc, fWi_bf, EMBc, fbi, gi_f, G3, Bc * Sc, G3, EMBc, 0);
  launch_gemm_big<0>(stream, f_bf, EMBc, bWi_bf, EMBc, bbi, gi_b, G3, Bc * Sc, G3, EMBc, 0);

  // ---- 5) encoder recurrence (fwd t=0..S-1, bwd t=S-1..0) ----
  const int pwBlocks = (Bc * HIDc + 255) / 256;
  for (int t = 0; t < Sc; ++t) {
    // forward direction
    launch_gemm_small(stream, h_f_bf, HIDc, fWh_bf, HIDc, fbh, gh_buf, nullptr, G3, Bc, G3, HIDc, 0);
    gru_enc_pw_k<<<pwBlocks, 256, 0, stream>>>(gi_f, t, gh_buf, h_f, h_f, h_f_bf, s_bf, 0);
    // backward direction
    int tb = Sc - 1 - t;
    launch_gemm_small(stream, h_b_bf, HIDc, bWh_bf, HIDc, bbh, gh_buf, nullptr, G3, Bc, G3, HIDc, 0);
    gru_enc_pw_k<<<pwBlocks, 256, 0, stream>>>(gi_b, tb, gh_buf, h_b, h_b, h_b_bf, s_bf, HIDc);
  }

  // ---- 6) proj_key = s @ att_k.W^T + b (big GEMM) ----
  launch_gemm_big<0>(stream, s_bf, 2 * HIDc, akW_bf, 2 * HIDc, akB, pk, ATTc,
                     Bc * Sc, ATTc, 2 * HIDc, 0);

  // ---- 7) decoder loop ----
  const int decPwBlocks = (Bc * TSZ + 255) / 256;
  for (int l = 0; l < Lc; ++l) {
    // q = t_prev @ att_q.W^T + b
    launch_gemm_small(stream, t_bf, TSZ, aqW_bf, TSZ, aqB, q_buf, nullptr, ATTc, Bc, ATTc, TSZ, 0);
    // scores (one wave per (b,s))
    attn_scores_k<<<(Bc * Sc * 32 + 255) / 256, 256, 0, stream>>>(q_buf, pk, avW, avB,
                                                                  x_mask, scores);
    // softmax + context + concat e_j -> h_dec
    attn_ctx_k<<<Bc, 256, 0, stream>>>(scores, s_bf, e_bf, hdec_f, hdec_bf, l);
    // GRU gates: gi = t_prev @ Wi^T, gh = h_dec @ Wh^T
    launch_gemm_small(stream, t_bf, TSZ, dWi_bf, TSZ, dbi, gi_dec, nullptr, D3, Bc, D3, TSZ, 0);
    launch_gemm_small(stream, hdec_bf, TSZ, dWh_bf, TSZ, dbh, gh_dec, nullptr, D3, Bc, D3, TSZ, 0);
    gru_dec_pw_k<<<decPwBlocks, 256, 0, stream>>>(gi_dec, gh_dec, hdec_f, t_bf, X_all, l);
  }

  // ---- 8) hoisted output projection: one giant GEMM (2048 x 32000 x 1536) ----
  launch_gemm_big<1>(stream, X_all, OUTK, oW_bf, OUTK, outB,
                     out, VOCc, Lc * Bc, VOCc, OUTK, 0);
}
